// TGN_6339371729529
// MI455X (gfx1250) — matrix operand order repaired and checked by reading, not compiled, not run on hardware
//
#include <hip/hip_runtime.h>
#include <hip/hip_bf16.h>
#include <stdint.h>

typedef __attribute__((ext_vector_type(16))) _Float16     v16h;
typedef __attribute__((ext_vector_type(8)))  _Float16     v8h;
typedef __attribute__((ext_vector_type(8)))  float        v8f;
typedef __attribute__((ext_vector_type(4)))  unsigned int u32x4;
typedef __attribute__((ext_vector_type(8)))  int          i32x8;
typedef __attribute__((ext_vector_type(4)))  int          i32x4;

// Problem constants (match reference)
constexpr int N_   = 100000;   // divisible by 16
constexpr int E_   = 400000;   // divisible by 16
constexpr int B_   = 4096;
constexpr int TD_  = 100;
constexpr int MSG_ = 100;
constexpr int ED_  = 200;      // TD+MSG
constexpr int EDP_ = 224;      // ED padded to multiple of 32 (WMMA K)
constexpr int K2_  = 256;      // link predictor K (2*EMB)

// ---- ordered-int float max helpers (monotonic int encoding) ----
__device__ __forceinline__ int f2ord(float f) {
    int i = __float_as_int(f);
    return (i >= 0) ? i : (i ^ 0x7fffffff);
}
__device__ __forceinline__ float ord2f(int i) {
    return __int_as_float((i >= 0) ? i : (i ^ 0x7fffffff));
}

// ---- A-fragment loader: 16-bit A 16x32 layout ----
// lane row m = lane&15; hs=lane>>4 selects K chunks: K(e)=kb+16*hs+e
__device__ __forceinline__ v16h load_a_frag(const _Float16* row, int kb, int hs) {
    const _Float16* ap = row + kb + 16 * hs;
    v8h lo = *(const v8h*)ap;
    v8h hi = *(const v8h*)(ap + 8);
    return __builtin_shufflevector(lo, hi, 0,1,2,3,4,5,6,7,8,9,10,11,12,13,14,15);
}
// ---- B-fragment: lane col n = lane&15, K = kb + 16*hs + e ; WT[col][K] f16 ----
__device__ __forceinline__ v16h load_b_frag(const _Float16* WT, int col, int kdim, int kb, int hs) {
    return *(const v16h*)(WT + (size_t)col * kdim + kb + 16 * hs);
}

// ---- TDM gather: 8 rows of 128 f16 from base[idx[i]*128 ..] into LDS ----
// D# per CDNA5 ISA 8.3-8.7: gather_mode=1, 32-bit indices, data_size=2B,
// tile_dim0=128, tile_dim1=8 (#indices), tensor_dim0_stride=128.
__device__ __forceinline__ void tdm_gather8(uint32_t lds_addr, const _Float16* base,
                                            i32x4 idxlo, i32x4 idxhi) {
    uint64_t ga = (uint64_t)(uintptr_t)base;
    u32x4 g0;
    g0.x = 0xC0000001u;                                   // count=1|gather_mode|idx32
    g0.y = lds_addr;                                      // lds_addr[31:0]
    g0.z = (uint32_t)ga;                                  // global_addr[31:0]
    g0.w = (uint32_t)((ga >> 32) & 0x1FFFFFFu) | (2u << 30); // addr[56:32]|type=2
    i32x8 g1;
    g1[0] = (1 << 16);                                    // data_size=2B, no mask
    g1[1] = (128 << 16);                                  // tensor_dim0[15:0]=128
    g1[2] = ((N_ & 0xFFFF) << 16);                        // tensor_dim1[15:0]
    g1[3] = (N_ >> 16) | (128 << 16);                     // tensor_dim1 hi | tile_dim0
    g1[4] = 8;                                            // tile_dim1 = #indices
    g1[5] = 128;                                          // tensor_dim0_stride lo
    g1[6] = 0;
    g1[7] = 0;
#if defined(__clang_major__) && (__clang_major__ >= 23)
    i32x8 zero8 = {};
    __builtin_amdgcn_tensor_load_to_lds(g0, g1, idxlo, idxhi, zero8, 0);
#else
    __builtin_amdgcn_tensor_load_to_lds(g0, g1, idxlo, idxhi, 0);
#endif
}

// =============== K0: weight convert/transpose + zero init ===============
__global__ void __launch_bounds__(256) prep_kernel(
    const float* Wq, const float* Wk, const float* Wv, const float* Wskip,
    const float* We, const float* Wh,
    const float* bq, const float* bk, const float* bv, const float* bskip,
    _Float16* WallT, _Float16* WeT, _Float16* WhT, float* bias_all,
    int* maxbits, float* denom, float* dout)
{
    constexpr int S0 = 512 * 128;          // WallT [col512][k128]
    constexpr int S1 = 128 * EDP_;         // WeT   [col128][k224]
    constexpr int S2 = 128 * K2_;          // WhT   [col128][k256]
    constexpr int S3 = 512;                // bias_all
    constexpr int S4 = 2 * N_;             // denom
    constexpr int S5 = 2;                  // maxbits
    constexpr int S6 = 1;                  // count slot
    constexpr int S7 = 2 * B_;             // y_true
    int idx = blockIdx.x * 256 + threadIdx.x;
    if (idx < S0) {
        int n = idx >> 7, k = idx & 127;
        int which = n >> 7, c = n & 127;
        const float* W = (which == 0) ? Wq : (which == 1) ? Wk : (which == 2) ? Wv : Wskip;
        WallT[idx] = (_Float16)W[k * 128 + c];
        return;
    }
    idx -= S0;
    if (idx < S1) {
        int n = idx / EDP_, k = idx % EDP_;
        WeT[idx] = (_Float16)((k < ED_) ? We[k * 128 + n] : 0.0f);
        return;
    }
    idx -= S1;
    if (idx < S2) {
        int n = idx >> 8, k = idx & 255;
        WhT[idx] = (_Float16)Wh[k * 128 + n];
        return;
    }
    idx -= S2;
    if (idx < S3) {
        float v = (idx < 128) ? bq[idx] : (idx < 256) ? bk[idx - 128]
                : (idx < 384) ? bv[idx - 256] : bskip[idx - 384];
        bias_all[idx] = v;
        return;
    }
    idx -= S3;
    if (idx < S4) { denom[idx] = 0.0f; return; }
    idx -= S4;
    if (idx < S5) { maxbits[idx] = f2ord(-3.402823466e38f); return; }
    idx -= S5;
    if (idx < S6) { dout[0] = 0.0f; return; }
    idx -= S6;
    if (idx < S7) { dout[1 + idx] = (idx < B_) ? 1.0f : 0.0f; return; }
}

// ---- uniform-select tile store for node projections ----
__device__ __forceinline__ void store_proj(int tile, int col, int row0, int hs, v8f acc,
    const float* bias_all, _Float16* qf, _Float16* kf, _Float16* vf, float* z)
{
    float bia = bias_all[col];
    int sel = tile >> 3;                 // wave-uniform: 0:q 1:k 2:v 3:z
    int c0 = col & 127;
    size_t base = (size_t)(row0 + 8 * hs) * 128 + c0;
    if (sel == 3) {
        #pragma unroll
        for (int r = 0; r < 8; ++r) z[base + (size_t)r * 128] = acc[r] + bia;
    } else {
        _Float16* dst = (sel == 0) ? qf : (sel == 1) ? kf : vf;
        #pragma unroll
        for (int r = 0; r < 8; ++r) dst[base + (size_t)r * 128] = (_Float16)(acc[r] + bia);
    }
}

// =============== K1: fused node projections  [N,128] x [128,512] ===============
__global__ void __launch_bounds__(256) node_proj_kernel(
    const float* __restrict__ x, const _Float16* __restrict__ WallT,
    const float* __restrict__ bias_all,
    _Float16* __restrict__ qf, _Float16* __restrict__ kf, _Float16* __restrict__ vf,
    float* __restrict__ z)
{
    __shared__ alignas(16) _Float16 A[16][128];
    int row0 = blockIdx.x * 16;
    int tid = threadIdx.x;
    for (int i = tid; i < 16 * 128; i += 256) {
        int r = i >> 7, c = i & 127;
        A[r][c] = (_Float16)x[(size_t)(row0 + r) * 128 + c];
    }
    __syncthreads();
    int wave = tid >> 5, lane = tid & 31;
    int hs = lane >> 4, nn = lane & 15;
    #pragma unroll
    for (int p = 0; p < 2; ++p) {
        int tileA = wave * 4 + p * 2;    // two tiles share the A fragment
        int colA = tileA * 16 + nn;
        int colB = colA + 16;
        v8f acc0 = {}, acc1 = {};
        #pragma unroll
        for (int kb = 0; kb < 128; kb += 32) {
            v16h a  = load_a_frag(&A[nn][0], kb, hs);
            v16h b0 = load_b_frag(WallT, colA, 128, kb, hs);
            v16h b1 = load_b_frag(WallT, colB, 128, kb, hs);
            acc0 = __builtin_amdgcn_wmma_f32_16x16x32_f16(false, a, false, b0,
                                                          (short)0, acc0, false, false);
            acc1 = __builtin_amdgcn_wmma_f32_16x16x32_f16(false, a, false, b1,
                                                          (short)0, acc1, false, false);
        }
        store_proj(tileA,     colA, row0, hs, acc0, bias_all, qf, kf, vf, z);
        store_proj(tileA + 1, colB, row0, hs, acc1, bias_all, qf, kf, vf, z);
    }
}

// =============== K2: per-edge: TDM gather + time-enc + ea GEMM + alpha + (v+ea) ===============
__global__ void __launch_bounds__(256) edge_kernel(
    const int* __restrict__ edge_index, const int* __restrict__ edge_t,
    const int* __restrict__ last_update, const float* __restrict__ edge_msg,
    const float* __restrict__ Wt, const float* __restrict__ bt,
    const _Float16* __restrict__ WeT,
    const _Float16* __restrict__ qf, const _Float16* __restrict__ kf,
    const _Float16* __restrict__ vf,
    float* __restrict__ alpha_out, _Float16* __restrict__ vpe, int* __restrict__ maxbits)
{
    __shared__ alignas(16) _Float16 A[16][EDP_];
    __shared__ alignas(16) _Float16 Q[16][128];
    __shared__ alignas(16) _Float16 Kt[16][128];
    __shared__ alignas(16) _Float16 V[16][128];
    __shared__ alignas(16) _Float16 P[16][128];
    __shared__ float AL[16][2];
    __shared__ int SRC[16];

    int e0 = blockIdx.x * 16;
    int tid = threadIdx.x;
    if (tid < 16) SRC[tid] = edge_index[e0 + tid];
    if (tid < 32) AL[tid >> 1][tid & 1] = 0.0f;

    // wave 0 only (TDM ignores EXEC; uniform branch skips other waves entirely):
    // async row-gathers of q[dst], k[src], v[src] into LDS via Tensor Data Mover
    if (tid < 32) {
        i32x4 dlo, dhi, d2lo, d2hi, slo, shi, s2lo, s2hi;
        #pragma unroll
        for (int j = 0; j < 4; ++j) {
            dlo[j]  = __builtin_amdgcn_readfirstlane(edge_index[E_ + e0 + j]);
            dhi[j]  = __builtin_amdgcn_readfirstlane(edge_index[E_ + e0 + 4 + j]);
            d2lo[j] = __builtin_amdgcn_readfirstlane(edge_index[E_ + e0 + 8 + j]);
            d2hi[j] = __builtin_amdgcn_readfirstlane(edge_index[E_ + e0 + 12 + j]);
            slo[j]  = __builtin_amdgcn_readfirstlane(edge_index[e0 + j]);
            shi[j]  = __builtin_amdgcn_readfirstlane(edge_index[e0 + 4 + j]);
            s2lo[j] = __builtin_amdgcn_readfirstlane(edge_index[e0 + 8 + j]);
            s2hi[j] = __builtin_amdgcn_readfirstlane(edge_index[e0 + 12 + j]);
        }
        tdm_gather8((uint32_t)(uintptr_t)&Q[0][0],  qf, dlo,  dhi);
        tdm_gather8((uint32_t)(uintptr_t)&Q[8][0],  qf, d2lo, d2hi);
        tdm_gather8((uint32_t)(uintptr_t)&Kt[0][0], kf, slo,  shi);
        tdm_gather8((uint32_t)(uintptr_t)&Kt[8][0], kf, s2lo, s2hi);
        tdm_gather8((uint32_t)(uintptr_t)&V[0][0],  vf, slo,  shi);
        tdm_gather8((uint32_t)(uintptr_t)&V[8][0],  vf, s2lo, s2hi);
    }
    __syncthreads();   // SRC visible for edge_attr build (TDM still in flight)

    // prefetch next block's edge messages into cache while we compute
    if (tid < 16 && e0 + 16 < E_) {
        __builtin_prefetch(edge_msg + (size_t)(e0 + 16) * MSG_ + tid * 25, 0, 1);
    }

    // build edge_attr rows: [cos-time-enc(100) | edge_msg(100) | zero-pad(24)]
    for (int i = tid; i < 16 * EDP_; i += 256) {
        int r = i / EDP_, c = i % EDP_;
        int e = e0 + r;
        float v = 0.0f;
        if (c < TD_) {
            float rt = (float)(last_update[SRC[r]] - edge_t[e]);
            v = __cosf(rt * Wt[c] + bt[c]);
        } else if (c < ED_) {
            v = edge_msg[(size_t)e * MSG_ + (c - TD_)];
        }
        A[r][c] = (_Float16)v;
    }
    if (tid < 32) __builtin_amdgcn_s_wait_tensorcnt(0);   // wave 0 TDM complete
    __syncthreads();

    int wave = tid >> 5, lane = tid & 31;
    int hs = lane >> 4, nn = lane & 15;
    int col = wave * 16 + nn;            // 8 waves -> 128 cols
    int head = col >> 6;
    v8f acc = {};
    #pragma unroll
    for (int kb = 0; kb < EDP_; kb += 32) {
        v16h a = load_a_frag(&A[nn][0], kb, hs);
        v16h b = load_b_frag(WeT, col, EDP_, kb, hs);
        acc = __builtin_amdgcn_wmma_f32_16x16x32_f16(false, a, false, b,
                                                     (short)0, acc, false, false);
    }
    #pragma unroll
    for (int r = 0; r < 8; ++r) {
        int m = r + 8 * hs;
        float ea = acc[r];
        float part = (float)Q[m][col] * ((float)Kt[m][col] + ea);
        part += __shfl_xor(part, 1, 16);             // reduce over 16 cols
        part += __shfl_xor(part, 2, 16);
        part += __shfl_xor(part, 4, 16);
        part += __shfl_xor(part, 8, 16);
        if (nn == 0) atomicAdd(&AL[m][head], part);  // 4 tiles/head accumulate
        P[m][col] = (_Float16)((float)V[m][col] + ea);
    }
    __syncthreads();

    // coalesced write of v+ea
    for (int i = tid; i < 16 * 64; i += 256) {
        int r = i >> 6, c = i & 63;
        ((uint32_t*)(vpe + (size_t)(e0 + r) * 128))[c] = ((const uint32_t*)&P[r][0])[c];
    }
    // per-edge alpha store (scaled by 1/sqrt(C))
    if (tid < 32) {
        int r = tid >> 1, h = tid & 1;
        alpha_out[(size_t)(e0 + r) * 2 + h] = AL[r][h] * 0.125f;
    }
    // block-level per-head max -> single global atomicMax per head
    if (tid < 2) {
        float mx = -3.402823466e38f;
        for (int r = 0; r < 16; ++r) mx = fmaxf(mx, AL[r][tid] * 0.125f);
        atomicMax(&maxbits[tid], f2ord(mx));
    }
}

// =============== K3: ex = exp(alpha - max); denom[dst] += ex ===============
__global__ void __launch_bounds__(256) ex_denom_kernel(
    const int* __restrict__ edge_index, const int* __restrict__ maxbits,
    float* __restrict__ alpha_ex, float* __restrict__ denom)
{
    int i = blockIdx.x * 256 + threadIdx.x;    // over E*2
    if (i >= E_ * 2) return;
    int e = i >> 1, h = i & 1;
    float ex = __expf(alpha_ex[i] - ord2f(maxbits[h]));
    alpha_ex[i] = ex;                           // in-place (rewritten by K2 each call)
    int d = edge_index[E_ + e];
    atomicAdd(&denom[(size_t)d * 2 + h], ex);
}

// =============== K4: z[dst] += (v+ea) * softmax ===============
__global__ void __launch_bounds__(256) scatter_kernel(
    const int* __restrict__ edge_index, const float* __restrict__ ex,
    const float* __restrict__ denom, const _Float16* __restrict__ vpe,
    float* __restrict__ z)
{
    int e = blockIdx.x * 2 + (threadIdx.x >> 7);
    int c = threadIdx.x & 127;
    int d = edge_index[E_ + e];
    int h = c >> 6;
    float a = ex[(size_t)e * 2 + h] / (denom[(size_t)d * 2 + h] + 1e-16f);
    float m = (float)vpe[(size_t)e * 128 + c] * a;
    atomicAdd(&z[(size_t)d * 128 + c], m);
}

// =============== K5: link predictor (8192 x 256 x 128 WMMA + fused head) ===============
__global__ void __launch_bounds__(256) linkpred_kernel(
    const float* __restrict__ z, const int* __restrict__ node_idx,
    const _Float16* __restrict__ WhT, const float* __restrict__ bh,
    const float* __restrict__ Wf, const float* __restrict__ bf,
    float* __restrict__ ypred)
{
    __shared__ alignas(16) _Float16 A[16][K2_];
    __shared__ float RED[16];
    __shared__ int NS[16], ND[16], MSK[16];
    int row0 = blockIdx.x * 16;     // rows over 2B
    int tid = threadIdx.x;
    if (tid < 16) {
        RED[tid] = 0.0f;
        int i = row0 + tid;
        int s, d;
        if (i < B_) { s = node_idx[i];        d = node_idx[B_ + i]; }
        else        { int j = i - B_; s = node_idx[j]; d = node_idx[2 * B_ + j]; }
        NS[tid] = s; ND[tid] = d;
        MSK[tid] = (s < N_ - 1) ? 1 : 0;
    }
    __syncthreads();
    for (int i = tid; i < 16 * K2_; i += 256) {
        int r = i >> 8, c = i & 255;
        const float* row = (c < 128) ? (z + (size_t)NS[r] * 128) : (z + (size_t)ND[r] * 128);
        A[r][c] = (_Float16)row[c & 127];
    }
    __syncthreads();
    int wave = tid >> 5, lane = tid & 31;
    int hs = lane >> 4, nn = lane & 15;
    int col = wave * 16 + nn;       // 8 waves -> 128 hidden cols
    v8f acc = {};
    #pragma unroll
    for (int kb = 0; kb < K2_; kb += 32) {
        v16h a = load_a_frag(&A[nn][0], kb, hs);
        v16h b = load_b_frag(WhT, col, K2_, kb, hs);
        acc = __builtin_amdgcn_wmma_f32_16x16x32_f16(false, a, false, b,
                                                     (short)0, acc, false, false);
    }
    float bhc = bh[col], wfc = Wf[col];
    #pragma unroll
    for (int r = 0; r < 8; ++r) {
        int m = r + 8 * hs;
        float hv = fmaxf(acc[r] + bhc, 0.0f) * wfc;  // ReLU then * Wf[col]
        hv += __shfl_xor(hv, 1, 16);
        hv += __shfl_xor(hv, 2, 16);
        hv += __shfl_xor(hv, 4, 16);
        hv += __shfl_xor(hv, 8, 16);
        if (nn == 0) atomicAdd(&RED[m], hv);
    }
    __syncthreads();
    if (tid < 16) {
        float p = (RED[tid] + bf[0]) * (float)MSK[tid];   // mask before sigmoid (ref)
        ypred[row0 + tid] = 1.0f / (1.0f + __expf(-p));
    }
}

// =============== K6: count = sum(node_idx[0] < N-1) ===============
__global__ void __launch_bounds__(256) count_kernel(
    const int* __restrict__ node_idx, float* __restrict__ out0)
{
    __shared__ int s;
    if (threadIdx.x == 0) s = 0;
    __syncthreads();
    int i = blockIdx.x * 256 + threadIdx.x;
    int v = (i < B_ && node_idx[i] < N_ - 1) ? 1 : 0;
    atomicAdd(&s, v);
    __syncthreads();
    if (threadIdx.x == 0) atomicAdd(out0, (float)s);
}

extern "C" void kernel_launch(void* const* d_in, const int* in_sizes, int n_in,
                              void* d_out, int out_size, void* d_ws, size_t ws_size,
                              hipStream_t stream) {
    const float* x           = (const float*)d_in[0];
    const int*   last_update = (const int*)  d_in[1];
    const int*   edge_index  = (const int*)  d_in[2];
    const int*   edge_t      = (const int*)  d_in[3];
    const float* edge_msg    = (const float*)d_in[4];
    const int*   node_idx    = (const int*)  d_in[5];
    const float* Wq = (const float*)d_in[6];  const float* bq = (const float*)d_in[7];
    const float* Wk = (const float*)d_in[8];  const float* bk = (const float*)d_in[9];
    const float* Wv = (const float*)d_in[10]; const float* bv = (const float*)d_in[11];
    const float* We = (const float*)d_in[12];
    const float* Wskip = (const float*)d_in[13]; const float* bskip = (const float*)d_in[14];
    const float* Wt = (const float*)d_in[15]; const float* bt = (const float*)d_in[16];
    const float* Wh = (const float*)d_in[17]; const float* bh = (const float*)d_in[18];
    const float* Wf = (const float*)d_in[19]; const float* bf = (const float*)d_in[20];
    float* out = (float*)d_out;

    char* ws = (char*)d_ws;
    size_t off = 0;
    auto take = [&](size_t bytes) -> char* {
        char* p = ws + off;
        off = (off + bytes + 255) & ~(size_t)255;
        return p;
    };
    _Float16* WallT    = (_Float16*)take((size_t)512 * 128 * 2);
    _Float16* WeT      = (_Float16*)take((size_t)128 * EDP_ * 2);
    _Float16* WhT      = (_Float16*)take((size_t)128 * K2_ * 2);
    float*    bias_all = (float*)   take((size_t)512 * 4);
    int*      maxbits  = (int*)     take((size_t)2 * 4);
    float*    denom    = (float*)   take((size_t)N_ * 2 * 4);
    _Float16* qf       = (_Float16*)take((size_t)N_ * 128 * 2);
    _Float16* kf       = (_Float16*)take((size_t)N_ * 128 * 2);
    _Float16* vf       = (_Float16*)take((size_t)N_ * 128 * 2);
    float*    z        = (float*)   take((size_t)N_ * 128 * 4);
    float*    alpha    = (float*)   take((size_t)E_ * 2 * 4);
    _Float16* vpe      = (_Float16*)take((size_t)E_ * 128 * 2);

    constexpr int PREP_TOT = 512*128 + 128*EDP_ + 128*K2_ + 512 + 2*N_ + 2 + 1 + 2*B_;
    prep_kernel<<<(PREP_TOT + 255) / 256, 256, 0, stream>>>(
        Wq, Wk, Wv, Wskip, We, Wh, bq, bk, bv, bskip,
        WallT, WeT, WhT, bias_all, maxbits, denom, out);

    node_proj_kernel<<<N_ / 16, 256, 0, stream>>>(
        x, WallT, bias_all, qf, kf, vf, z);

    edge_kernel<<<E_ / 16, 256, 0, stream>>>(
        edge_index, edge_t, last_update, edge_msg, Wt, bt, WeT,
        qf, kf, vf, alpha, vpe, maxbits);

    ex_denom_kernel<<<(E_ * 2 + 255) / 256, 256, 0, stream>>>(
        edge_index, maxbits, alpha, denom);

    scatter_kernel<<<E_ / 2, 256, 0, stream>>>(
        edge_index, alpha, denom, vpe, z);

    linkpred_kernel<<<(2 * B_) / 16, 256, 0, stream>>>(
        z, node_idx, WhT, bh, Wf, bf, out + 1 + 2 * B_);

    count_kernel<<<(B_ + 255) / 256, 256, 0, stream>>>(node_idx, out);
}
